// LogicTreeNetCIFAR10_14087492731145
// MI455X (gfx1250) — matrix operand order repaired and checked by reading, not compiled, not run on hardware
//
#include <hip/hip_runtime.h>
#include <hip/hip_bf16.h>
#include <math.h>

// ---------------------------------------------------------------------------
// LogicTreeNet CIFAR-10 forward, MI455X (gfx1250, wave32).
//
// ~24M gate evaluations (~0.7 GFLOP), ~60MB of intermediates -> L2/latency
// bound; every per-image activation row fits in the 320KB WGP LDS. Design:
//  - stage each layer's per-image input in LDS via CDNA5 async-to-LDS DMA
//    (global_load_async_to_lds_b128 + s_wait_asynccnt), gather from LDS
//  - fuse or_pool into the tree-conv kernels (4 conv px -> 1 pooled value)
//  - ping-pong activations through two L2-resident workspace arenas
//  - the one real GEMM, softmax@GATE_COEF (R,16)x(16,4) over 83552 rows,
//    runs on the matrix unit with v_wmma_f32_16x16x32_f16
// ---------------------------------------------------------------------------

typedef __attribute__((ext_vector_type(16))) _Float16 v16h;
typedef __attribute__((ext_vector_type(8)))  float    v8f;

__device__ __constant__ float GC[16][4] = {
    {0,0,0,0},{0,0,0,1},{0,1,0,-1},{0,1,0,0},{0,0,1,-1},{0,0,1,0},{0,1,1,-2},{0,1,1,-1},
    {1,-1,-1,1},{1,-1,-1,2},{1,0,-1,0},{1,0,-1,1},{1,-1,0,0},{1,-1,0,1},{1,0,0,-1},{1,0,0,0}};

__device__ __forceinline__ float gate4(float a, float b, float w0, float w1, float w2, float w3) {
    return fmaf(w3, a * b, fmaf(w2, b, fmaf(w1, a, w0)));
}

// ---------------------------------------------------------------------------
// Async memory->LDS staging (CDNA5 Tensor/async path, ASYNCcnt-tracked).
// Copies nf4 float4's from src (contiguous) into dynamic LDS.
// NOTE: callers have NO static __shared__, so dynamic LDS starts at byte
// offset 0 and the raw LDS offset of element i is exactly i*16.
// nf4 is always a multiple of blockDim.x here -> full-EXEC, no tail.
// ---------------------------------------------------------------------------
__device__ __forceinline__ void lds_stage_async_f4(const float* __restrict__ src,
                                                   int nf4, int tid, int nthreads)
{
    for (int i = tid; i < nf4; i += nthreads) {
        unsigned loff = (unsigned)i * 16u;   // LDS byte offset (dynamic base = 0)
        unsigned goff = (unsigned)i * 16u;   // global byte offset from src
        asm volatile("global_load_async_to_lds_b128 %0, %1, %2 offset:0"
                     :
                     : "v"(loff), "v"(goff), "s"(src)
                     : "memory");
    }
    asm volatile("s_wait_asynccnt 0x0" ::: "memory");
}

// ---------------------------------------------------------------------------
// Gate-mix GEMM: m[r, 0:4] = softmax(logits[r, 0:16]) @ GATE_COEF
// One wave32 per 16 rows (rows is always a multiple of 16). Lanes l and l+16
// both own row (l&15), so each lane already holds the 8 K-values its
// A-fragment slot needs (A 16x32 f16: lanes 0-15 -> K=0..7, lanes 16-31 ->
// K=8..15, K=16..31 zero). B 32x16: element e<8 = GATE_COEF[koff+e][n]
// for n=(lane&15)<4 else 0; elements 8-15 (K>=16) zero. C = 0. D: lane holds column
// (lane&15), VGPR r is row r + 8*(lane>=16); lanes with column<4 store m.
// ---------------------------------------------------------------------------
__global__ __launch_bounds__(256) void mix_wmma_kernel(const float* __restrict__ logits,
                                                       float* __restrict__ m_out, int rows)
{
    const int lane = threadIdx.x & 31;
    const int wave = (int)(blockIdx.x * (blockDim.x >> 5) + (threadIdx.x >> 5));
    const int base = wave * 16;
    if (base >= rows) return;                 // wave-uniform: EXEC stays all-ones

    const int r = base + (lane & 15);
    const float* zr = logits + (size_t)r * 16;
    float z[16];
    #pragma unroll
    for (int k = 0; k < 16; ++k) z[k] = zr[k];
    float mx = z[0];
    #pragma unroll
    for (int k = 1; k < 16; ++k) mx = fmaxf(mx, z[k]);
    float p[16], s = 0.0f;
    #pragma unroll
    for (int k = 0; k < 16; ++k) { p[k] = __expf(z[k] - mx); s += p[k]; }
    const float inv = 1.0f / s;

    v16h a, b;
    #pragma unroll
    for (int e = 0; e < 16; ++e) { a[e] = (_Float16)0.0f; b[e] = (_Float16)0.0f; }
    const int koff = (lane < 16) ? 0 : 8;
    #pragma unroll
    for (int e = 0; e < 8; ++e) a[e] = (_Float16)(p[koff + e] * inv);
    const int n = lane & 15;
    if (n < 4) {
        #pragma unroll
        for (int e = 0; e < 8; ++e) b[e] = (_Float16)GC[koff + e][n];
    }
    v8f c;
    #pragma unroll
    for (int e = 0; e < 8; ++e) c[e] = 0.0f;

    v8f d = __builtin_amdgcn_wmma_f32_16x16x32_f16(false, a, false, b, (short)0, c, false, false);

    if (n < 4) {                               // rows % 16 == 0: no row guard needed
        const int rbase = base + ((lane >> 4) << 3);
        #pragma unroll
        for (int rr = 0; rr < 8; ++rr)
            m_out[(size_t)(rbase + rr) * 4 + n] = d[rr];
    }
}

// ---------------------------------------------------------------------------
// binarize: x (B,3,32,32) -> h (B,9,32,32); channel = t*3 + c, thresholds .25/.5/.75
// ---------------------------------------------------------------------------
__global__ __launch_bounds__(256) void binarize_kernel(const float* __restrict__ x,
                                                       float* __restrict__ h, int n)
{
    int i = blockIdx.x * blockDim.x + threadIdx.x;
    if (i >= n) return;
    const int hw = 32 * 32;
    int bimg = i / (3 * hw);
    int cch  = (i / hw) % 3;
    int pix  = i % hw;
    float v = x[i];
    const float th[3] = {0.25f, 0.5f, 0.75f};
    #pragma unroll
    for (int t = 0; t < 3; ++t)
        h[((size_t)bimg * 9 + t * 3 + cch) * hw + pix] = (v > th[t]) ? 1.0f : 0.0f;
}

// ---------------------------------------------------------------------------
// Fused tree_conv + or_pool. One WG per (image, O-block); whole C*H*W input
// image staged into dynamic LDS by async DMA; each thread computes one pooled
// cell: 4 conv pixels (8 LDS gathers + 7 gate FMAs each) -> max.
// Output (B,O,H/2,W/2). Tile float counts are multiples of 4*blockDim.
// ---------------------------------------------------------------------------
__global__ __launch_bounds__(256) void tree_conv_pool_kernel(
    const float* __restrict__ in, const int* __restrict__ leaf_idx,
    const float* __restrict__ m, float* __restrict__ out,
    int C, int H, int W, int O, int OBLK)
{
    extern __shared__ float lds[];
    const int nob = O / OBLK;
    const int bimg = blockIdx.x / nob;
    const int ob   = blockIdx.x % nob;

    const int tile = C * H * W;
    lds_stage_async_f4(in + (size_t)bimg * tile, tile >> 2, threadIdx.x, blockDim.x);
    __syncthreads();

    const int PH = H >> 1, PW = W >> 1;
    const int cells = OBLK * PH * PW;
    for (int t = threadIdx.x; t < cells; t += blockDim.x) {
        const int o  = ob * OBLK + t / (PH * PW);
        const int py = (t / PW) % PH;
        const int px = t % PW;

        int ch[8], dy[8], dx[8];
        const int* li = leaf_idx + o * 8;
        #pragma unroll
        for (int l = 0; l < 8; ++l) {
            int idx = li[l];
            ch[l] = idx / 9;
            int pos = idx % 9;
            dy[l] = pos / 3 - 1;
            dx[l] = pos % 3 - 1;
        }
        float w[7][4];
        const float* mo = m + (size_t)o * 28;
        #pragma unroll
        for (int nd = 0; nd < 7; ++nd)
            #pragma unroll
            for (int q = 0; q < 4; ++q) w[nd][q] = mo[nd * 4 + q];

        float best = -INFINITY;
        #pragma unroll
        for (int sy = 0; sy < 2; ++sy)
            #pragma unroll
            for (int sx = 0; sx < 2; ++sx) {
                const int y = 2 * py + sy, x = 2 * px + sx;
                float v[8];
                #pragma unroll
                for (int l = 0; l < 8; ++l) {
                    int yy = y + dy[l], xx = x + dx[l];
                    v[l] = (yy >= 0 && yy < H && xx >= 0 && xx < W)
                               ? lds[(ch[l] * H + yy) * W + xx] : 0.0f;
                }
                float t0 = gate4(v[0], v[1], w[0][0], w[0][1], w[0][2], w[0][3]);
                float t1 = gate4(v[2], v[3], w[1][0], w[1][1], w[1][2], w[1][3]);
                float t2 = gate4(v[4], v[5], w[2][0], w[2][1], w[2][2], w[2][3]);
                float t3 = gate4(v[6], v[7], w[3][0], w[3][1], w[3][2], w[3][3]);
                float u0 = gate4(t0, t1, w[4][0], w[4][1], w[4][2], w[4][3]);
                float u1 = gate4(t2, t3, w[5][0], w[5][1], w[5][2], w[5][3]);
                float r  = gate4(u0, u1, w[6][0], w[6][1], w[6][2], w[6][3]);
                best = fmaxf(best, r);
            }
        out[(((size_t)bimg * O + o) * PH + py) * PW + px] = best;
    }
}

// ---------------------------------------------------------------------------
// Logic layer: out[b,j] = mix(w[j]) applied to in[b,a[j]], in[b,b[j]].
// One WG per (image, j-block); the full Din-float input row is DMA'd into
// dynamic LDS (max 160KB for l2 — legal on CDNA5's 320KB WGP LDS).
// ---------------------------------------------------------------------------
__global__ __launch_bounds__(256) void logic_layer_kernel(
    const float* __restrict__ in, const int* __restrict__ aidx,
    const int* __restrict__ bidx, const float4* __restrict__ m,
    float* __restrict__ out, int Din, int Dout, int nj)
{
    extern __shared__ float lds[];
    const int bimg = blockIdx.x / nj;
    const int jb   = blockIdx.x % nj;
    lds_stage_async_f4(in + (size_t)bimg * Din, Din >> 2, threadIdx.x, blockDim.x);
    __syncthreads();

    const int per = Dout / nj;
    const int j0  = jb * per;
    float* dst = out + (size_t)bimg * Dout;
    for (int j = j0 + threadIdx.x; j < j0 + per; j += blockDim.x) {
        const float a = lds[aidx[j]];
        const float b = lds[bidx[j]];
        const float4 w = m[j];
        dst[j] = fmaf(w.w, a * b, fmaf(w.z, b, fmaf(w.y, a, w.x)));
    }
}

// ---------------------------------------------------------------------------
// group_sum: out[b,c] = sum_{i<1024} f3[b, c*1024 + i] / 100
// ---------------------------------------------------------------------------
__global__ __launch_bounds__(256) void group_sum_kernel(const float* __restrict__ in,
                                                        float* __restrict__ out, int per_class)
{
    __shared__ float red[256];
    const float* src = in + (size_t)blockIdx.x * per_class;
    float s = 0.0f;
    for (int i = threadIdx.x; i < per_class; i += blockDim.x) s += src[i];
    red[threadIdx.x] = s;
    __syncthreads();
    for (int off = 128; off >= 1; off >>= 1) {
        if (threadIdx.x < (unsigned)off) red[threadIdx.x] += red[threadIdx.x + off];
        __syncthreads();
    }
    if (threadIdx.x == 0) out[blockIdx.x] = red[0] / 100.0f;
}

// ---------------------------------------------------------------------------
// Host-side launch sequence
// ---------------------------------------------------------------------------
extern "C" void kernel_launch(void* const* d_in, const int* in_sizes, int n_in,
                              void* d_out, int out_size, void* d_ws, size_t ws_size,
                              hipStream_t stream)
{
    const float* x      = (const float*)d_in[0];
    const int*   c1_idx = (const int*)d_in[1];  const float* c1_w = (const float*)d_in[2];
    const int*   c2_idx = (const int*)d_in[3];  const float* c2_w = (const float*)d_in[4];
    const int*   c3_idx = (const int*)d_in[5];  const float* c3_w = (const float*)d_in[6];
    const int*   c4_idx = (const int*)d_in[7];  const float* c4_w = (const float*)d_in[8];
    const int*   l1_a = (const int*)d_in[9];   const int* l1_b = (const int*)d_in[10];
    const float* l1_w = (const float*)d_in[11];
    const int*   l2_a = (const int*)d_in[12];  const int* l2_b = (const int*)d_in[13];
    const float* l2_w = (const float*)d_in[14];
    const int*   l3_a = (const int*)d_in[15];  const int* l3_b = (const int*)d_in[16];
    const float* l3_w = (const float*)d_in[17];
    float* out = (float*)d_out;

    const int B = 128, K = 32;

    // --- workspace layout (floats); total ~32.8 MB, ping-pong arenas ---
    float* ws = (float*)d_ws;
    float* m_c1 = ws;                 // 224*4
    float* m_c2 = m_c1 + 224 * 4;     // 896*4
    float* m_c3 = m_c2 + 896 * 4;     // 3584*4
    float* m_c4 = m_c3 + 3584 * 4;    // 7168*4
    float* m_l1 = m_c4 + 7168 * 4;    // 40960*4
    float* m_l2 = m_l1 + 40960 * 4;   // 20480*4
    float* m_l3 = m_l2 + 20480 * 4;   // 10240*4
    float* A0   = m_l3 + 10240 * 4;   // arena0: 2,621,440 floats (h0,a2,a4,f2)
    float* A1   = A0 + 2621440;       // arena1: 5,242,880 floats (a1,a3,f1,f3)

    // --- gate-mix GEMMs on the matrix unit (rows all multiples of 16) ---
    auto mixgrid = [](int rows) { return (rows + 127) / 128; };
    hipLaunchKernelGGL(mix_wmma_kernel, dim3(mixgrid(224)),   dim3(256), 0, stream, c1_w, m_c1, 224);
    hipLaunchKernelGGL(mix_wmma_kernel, dim3(mixgrid(896)),   dim3(256), 0, stream, c2_w, m_c2, 896);
    hipLaunchKernelGGL(mix_wmma_kernel, dim3(mixgrid(3584)),  dim3(256), 0, stream, c3_w, m_c3, 3584);
    hipLaunchKernelGGL(mix_wmma_kernel, dim3(mixgrid(7168)),  dim3(256), 0, stream, c4_w, m_c4, 7168);
    hipLaunchKernelGGL(mix_wmma_kernel, dim3(mixgrid(40960)), dim3(256), 0, stream, l1_w, m_l1, 40960);
    hipLaunchKernelGGL(mix_wmma_kernel, dim3(mixgrid(20480)), dim3(256), 0, stream, l2_w, m_l2, 20480);
    hipLaunchKernelGGL(mix_wmma_kernel, dim3(mixgrid(10240)), dim3(256), 0, stream, l3_w, m_l3, 10240);

    // --- binarize: x -> h0 (arena0) ---
    {
        const int n = B * 3 * 32 * 32;
        hipLaunchKernelGGL(binarize_kernel, dim3((n + 255) / 256), dim3(256), 0, stream, x, A0, n);
    }

    // --- conv stack (async-DMA input tile in dynamic LDS, or_pool fused) ---
    // c1: (B,9,32,32)->(B,32,16,16)   LDS 36864B
    hipLaunchKernelGGL(tree_conv_pool_kernel, dim3(B * 2),  dim3(256), 9 * 32 * 32 * 4,  stream,
                       A0, c1_idx, m_c1, A1, 9, 32, 32, K, 16);
    // c2: (B,32,16,16)->(B,128,8,8)   LDS 32768B
    hipLaunchKernelGGL(tree_conv_pool_kernel, dim3(B * 4),  dim3(256), 32 * 16 * 16 * 4, stream,
                       A1, c2_idx, m_c2, A0, 32, 16, 16, 4 * K, 32);
    // c3: (B,128,8,8)->(B,512,4,4)    LDS 32768B
    hipLaunchKernelGGL(tree_conv_pool_kernel, dim3(B * 4),  dim3(256), 128 * 8 * 8 * 4,  stream,
                       A0, c3_idx, m_c3, A1, 128, 8, 8, 16 * K, 128);
    // c4: (B,512,4,4)->(B,1024,2,2)   LDS 32768B ; flatten = channel-major, matches reshape
    hipLaunchKernelGGL(tree_conv_pool_kernel, dim3(B * 4),  dim3(256), 512 * 4 * 4 * 4,  stream,
                       A1, c4_idx, m_c4, A0, 512, 4, 4, 32 * K, 256);

    // --- logic layers (full input row DMA'd to dynamic LDS) ---
    // l1: 4096 -> 40960   LDS 16KB
    hipLaunchKernelGGL(logic_layer_kernel, dim3(B * 8), dim3(256), 4096 * 4,  stream,
                       A0, l1_a, l1_b, (const float4*)m_l1, A1, 4096, 40960, 8);
    // l2: 40960 -> 20480  LDS 160KB (CDNA5 WGP LDS = 320KB)
    hipLaunchKernelGGL(logic_layer_kernel, dim3(B * 4), dim3(256), 40960 * 4, stream,
                       A1, l2_a, l2_b, (const float4*)m_l2, A0, 40960, 20480, 4);
    // l3: 20480 -> 10240  LDS 80KB
    hipLaunchKernelGGL(logic_layer_kernel, dim3(B * 4), dim3(256), 20480 * 4, stream,
                       A0, l3_a, l3_b, (const float4*)m_l3, A1, 20480, 10240, 4);

    // --- group sum: (B,10,1024) -> (B,10) ---
    hipLaunchKernelGGL(group_sum_kernel, dim3(B * 10), dim3(256), 0, stream, A1, out, 1024);

    (void)in_sizes; (void)n_in; (void)out_size; (void)ws_size;
}
